// EGNNModel_41755672052200
// MI455X (gfx1250) — matrix operand order repaired and checked by reading, not compiled, not run on hardware
//
#include <hip/hip_runtime.h>
#include <math.h>

typedef unsigned int u32;
typedef unsigned short u16;
typedef unsigned char u8;

typedef float v8f __attribute__((ext_vector_type(8)));
typedef u32 v8u __attribute__((ext_vector_type(8)));
typedef u32 v4u __attribute__((ext_vector_type(4)));
typedef __bf16 v16bf __attribute__((ext_vector_type(16)));

// ---------- helpers ----------
__device__ __forceinline__ u32 f2bf(float f) {
  u32 u = __builtin_bit_cast(u32, f);
  return (u + 0x7FFFu + ((u >> 16) & 1u)) >> 16;  // RNE
}
__device__ __forceinline__ u32 pack_bf16(float a, float b) {
  return f2bf(a) | (f2bf(b) << 16);
}
__device__ __forceinline__ float bf2f(u16 h) {
  u32 u = ((u32)h) << 16;
  return __builtin_bit_cast(float, u);
}
__device__ __forceinline__ float silu(float v) { return v / (1.0f + __expf(-v)); }

// A-fragment: 16x32 bf16 tile staged in LDS as packed-u32 rows (krowU32 u32 per row).
// ISA 7.12.2 layout: lane L (m=L&15, half=L>>4), VGPR v holds K pair
//   kv = (v<4 ? 2v : 16+2(v-4)) + 8*half  ->  u32 idx = kBase/2 + 4*half + (v<4 ? v : 8+v-4)
// i.e. two runs of 4 consecutive u32 -> two 16B LDS loads (ds_load_b128).
__device__ __forceinline__ v16bf load_a_frag(const u32* lds, int krowU32, int kBase, int lane) {
  const int m = lane & 15, half = lane >> 4;
  const u32* p = lds + m * krowU32 + (kBase >> 1) + half * 4;
  v4u lo = *(const v4u*)p;
  v4u hi = *(const v4u*)(p + 8);
  v8u a = __builtin_shufflevector(lo, hi, 0, 1, 2, 3, 4, 5, 6, 7);
  return __builtin_bit_cast(v16bf, a);
}
// B-fragment: pre-packed lane-major in global memory: 32 contiguous bytes per lane
// -> 2x global_load_b128, fully coalesced across the wave.
__device__ __forceinline__ v16bf load_b_frag(const u32* frag, int lane) {
  v8u b = *(const v8u*)(frag + lane * 8);
  return __builtin_bit_cast(v16bf, b);
}
__device__ __forceinline__ v8f wmma_bf16(v16bf A, v16bf B, v8f C) {
  return __builtin_amdgcn_wmma_f32_16x16x32_bf16(false, A, false, B, (short)0, C, false, false);
}

// ---------- weight packer: W[K x 128] f32 row-major -> bf16 fragment order ----------
// Fragment (kt,nt) occupies 256 u32 at frag + (kt*8+nt)*256, element [lane*8 + v].
// B 32x16 ISA layout: VGPR v, half h: K = kt*32 + 2(v&3) + 16(v>>2) + 8h (lo16), +1 (hi16); col = nt*16 + (lane&15).
__global__ __launch_bounds__(32) void pack_w_kernel(const float* __restrict__ W,
                                                    u32* __restrict__ out, int K) {
  const int lane = threadIdx.x;
  const int half = lane >> 4, n = lane & 15;
  const int nt = blockIdx.x & 7;
  const int col = nt * 16 + n;
  const int kt = blockIdx.x >> 3;
  (void)K;
#pragma unroll
  for (int v = 0; v < 8; ++v) {
    int k0 = kt * 32 + 2 * (v & 3) + 16 * (v >> 2) + 8 * half;
    out[((size_t)blockIdx.x * 32 + lane) * 8 + v] =
        pack_bf16(W[(size_t)k0 * 128 + col], W[(size_t)(k0 + 1) * 128 + col]);
  }
}

// ---------- fused edge kernel: gather -> edgeMLP -> coordMLP -> scatter ----------
__global__ __launch_bounds__(32) void egnn_edge_kernel(
    const float* __restrict__ h, const float* __restrict__ x, const int* __restrict__ esrc,
    const int* __restrict__ edst, const u32* __restrict__ fW1, const float* __restrict__ w1r,
    const float* __restrict__ eb1, const u32* __restrict__ fW2, const float* __restrict__ eb2,
    const u32* __restrict__ fC1, const float* __restrict__ cb1, const float* __restrict__ cw2,
    float* __restrict__ h_neigh, float* __restrict__ x_sum, float* __restrict__ deg, int din) {
  __shared__ u32 ldsA[16 * 128];   // [h_src | h_dst] bf16, up to 256 per row
  __shared__ u32 ldsB1[16 * 64];   // 16x128 bf16 scratch
  __shared__ u32 ldsB2[16 * 64];   // 16x128 bf16 scratch
  __shared__ float ldsRad[16];
  __shared__ float ldsXd[16][3];
  __shared__ int ldsSrc[16];
  __shared__ int ldsDst[16];

  const int lane = threadIdx.x;
  const int e0 = blockIdx.x * 16;

  if (lane < 16) {
    int e = e0 + lane;
    int s = esrc[e], d = edst[e];
    ldsSrc[lane] = s;
    ldsDst[lane] = d;
    float dx = x[3 * s + 0] - x[3 * d + 0];
    float dy = x[3 * s + 1] - x[3 * d + 1];
    float dz = x[3 * s + 2] - x[3 * d + 2];
    float r = dx * dx + dy * dy + dz * dz;
    float inv = 1.0f / (sqrtf(r) + 1e-30f);
    ldsRad[lane] = r;
    ldsXd[lane][0] = dx * inv;
    ldsXd[lane][1] = dy * inv;
    ldsXd[lane][2] = dz * inv;
  }
  __syncthreads();

  // stage A = [h[src] | h[dst]] as bf16; row = 2*din bf16 = din u32
  for (int idx = lane; idx < 16 * din; idx += 32) {
    int row = idx / din, p = idx % din;
    int c0 = 2 * p;
    float a, b;
    if (c0 < din) {
      const float* r = h + (size_t)ldsSrc[row] * din;
      a = r[c0];
      b = r[c0 + 1];
    } else {
      const float* r = h + (size_t)ldsDst[row] * din;
      a = r[c0 - din];
      b = r[c0 - din + 1];
    }
    ldsA[row * din + p] = pack_bf16(a, b);
  }
  __syncthreads();

  const int half = lane >> 4, nn = lane & 15;
  const int KT1 = (2 * din) >> 5;

  // hoist per-row values out of the N-tile loops (invariant across nt)
  float radv[8];
  int dstv[8];
#pragma unroll
  for (int v = 0; v < 8; ++v) {
    radv[v] = ldsRad[v + 8 * half];
    dstv[v] = ldsDst[v + 8 * half];
  }

  // GEMM1: hidden1 = silu([hs|hd]@W1ab + radial*w1row + b1)  -> ldsB1
  for (int nt = 0; nt < 8; ++nt) {
    int col = nt * 16 + nn;
    float bias = eb1[col], wr = w1r[col];
    v8f acc;
#pragma unroll
    for (int v = 0; v < 8; ++v) acc[v] = bias + radv[v] * wr;
    for (int kt = 0; kt < KT1; ++kt)
      acc = wmma_bf16(load_a_frag(ldsA, din, kt * 32, lane),
                      load_b_frag(fW1 + (size_t)(kt * 8 + nt) * 256, lane), acc);
#pragma unroll
    for (int v = 0; v < 8; ++v)
      ((u16*)ldsB1)[(v + 8 * half) * 128 + col] = (u16)f2bf(silu(acc[v]));
  }
  __syncthreads();

  // GEMM2: msg_h = silu(hidden1@W2 + b2); scatter into h_neigh, keep bf16 in ldsB2
  for (int nt = 0; nt < 8; ++nt) {
    int col = nt * 16 + nn;
    float bias = eb2[col];
    v8f acc;
#pragma unroll
    for (int v = 0; v < 8; ++v) acc[v] = bias;
    for (int kt = 0; kt < 4; ++kt)
      acc = wmma_bf16(load_a_frag(ldsB1, 64, kt * 32, lane),
                      load_b_frag(fW2 + (size_t)(kt * 8 + nt) * 256, lane), acc);
#pragma unroll
    for (int v = 0; v < 8; ++v) {
      int m = v + 8 * half;
      float s = silu(acc[v]);
      ((u16*)ldsB2)[m * 128 + col] = (u16)f2bf(s);
      unsafeAtomicAdd(&h_neigh[(size_t)dstv[v] * 128 + col], s);
    }
  }
  __syncthreads();

  // GEMM3: c1 = silu(msg_h@CW1 + cb1) -> ldsB1
  for (int nt = 0; nt < 8; ++nt) {
    int col = nt * 16 + nn;
    float bias = cb1[col];
    v8f acc;
#pragma unroll
    for (int v = 0; v < 8; ++v) acc[v] = bias;
    for (int kt = 0; kt < 4; ++kt)
      acc = wmma_bf16(load_a_frag(ldsB2, 64, kt * 32, lane),
                      load_b_frag(fC1 + (size_t)(kt * 8 + nt) * 256, lane), acc);
#pragma unroll
    for (int v = 0; v < 8; ++v)
      ((u16*)ldsB1)[(v + 8 * half) * 128 + col] = (u16)f2bf(silu(acc[v]));
  }
  __syncthreads();

  // coef = c1 @ cw2 ; scatter coord messages + degree.
  // split the 128-dot over the two lane halves, combine with a cross-half shuffle.
  {
    const int m = lane & 15;
    const u16* c1 = (const u16*)ldsB1 + m * 128 + half * 64;
    const float* w2 = cw2 + half * 64;
    float cf = 0.f;
    for (int j = 0; j < 64; ++j) cf += bf2f(c1[j]) * w2[j];
    cf += __shfl_xor(cf, 16, 32);
    if (lane < 16) {
      int d = ldsDst[lane];
      unsafeAtomicAdd(&x_sum[3 * d + 0], cf * ldsXd[lane][0]);
      unsafeAtomicAdd(&x_sum[3 * d + 1], cf * ldsXd[lane][1]);
      unsafeAtomicAdd(&x_sum[3 * d + 2], cf * ldsXd[lane][2]);
      unsafeAtomicAdd(&deg[d], 1.0f);
    }
  }
}

// ---------- node update kernel ----------
__global__ __launch_bounds__(32) void egnn_node_kernel(
    const float* __restrict__ h, const float* __restrict__ h_neigh, const float* __restrict__ x,
    const float* __restrict__ x_sum, const float* __restrict__ deg, const u32* __restrict__ fN1,
    const float* __restrict__ nb1, const u32* __restrict__ fN2, const float* __restrict__ nb2,
    float* __restrict__ h_out, float* __restrict__ x_out, int din) {
  __shared__ u32 ldsA[16 * 128];
  __shared__ u32 ldsB1[16 * 64];
  const int lane = threadIdx.x;
  const int n0 = blockIdx.x * 16;
  const int K = din + 128, Ku = K >> 1;

  for (int idx = lane; idx < 16 * Ku; idx += 32) {
    int row = idx / Ku, p = idx % Ku;
    int c0 = 2 * p;
    int node = n0 + row;
    float a, b;
    if (c0 < din) {
      const float* r = h + (size_t)node * din;
      a = r[c0];
      b = r[c0 + 1];
    } else {
      const float* r = h_neigh + (size_t)node * 128;
      a = r[c0 - din];
      b = r[c0 - din + 1];
    }
    ldsA[row * Ku + p] = pack_bf16(a, b);
  }
  __syncthreads();

  const int half = lane >> 4, nn = lane & 15;
  const int KT = K >> 5;
  for (int nt = 0; nt < 8; ++nt) {
    int col = nt * 16 + nn;
    float bias = nb1[col];
    v8f acc;
#pragma unroll
    for (int v = 0; v < 8; ++v) acc[v] = bias;
    for (int kt = 0; kt < KT; ++kt)
      acc = wmma_bf16(load_a_frag(ldsA, Ku, kt * 32, lane),
                      load_b_frag(fN1 + (size_t)(kt * 8 + nt) * 256, lane), acc);
#pragma unroll
    for (int v = 0; v < 8; ++v)
      ((u16*)ldsB1)[(v + 8 * half) * 128 + col] = (u16)f2bf(silu(acc[v]));
  }
  __syncthreads();

  for (int nt = 0; nt < 8; ++nt) {
    int col = nt * 16 + nn;
    float bias = nb2[col];
    v8f acc;
#pragma unroll
    for (int v = 0; v < 8; ++v) acc[v] = bias;
    for (int kt = 0; kt < 4; ++kt)
      acc = wmma_bf16(load_a_frag(ldsB1, 64, kt * 32, lane),
                      load_b_frag(fN2 + (size_t)(kt * 8 + nt) * 256, lane), acc);
#pragma unroll
    for (int v = 0; v < 8; ++v)
      h_out[(size_t)(n0 + v + 8 * half) * 128 + col] = acc[v];
  }

  if (lane < 16) {
    int node = n0 + lane;
    float d = deg[node];
#pragma unroll
    for (int k = 0; k < 3; ++k) {
      float xs = x_sum[3 * node + k];
      float xn = d > 0.f ? xs / fmaxf(d, 1.f) : 0.f;
      x_out[3 * node + k] = x[3 * node + k] + xn;
    }
  }
}

// ---------- final node MLP + sum-pool scatter ----------
__global__ __launch_bounds__(32) void egnn_final_node_kernel(
    const float* __restrict__ h, const int* __restrict__ gids, const u32* __restrict__ f0,
    const float* __restrict__ b0, const u32* __restrict__ f1, const float* __restrict__ b1,
    float* __restrict__ g) {
  __shared__ u32 ldsA[16 * 64];
  __shared__ u32 ldsB1[16 * 64];
  __shared__ int ldsG[16];
  const int lane = threadIdx.x;
  const int n0 = blockIdx.x * 16;
  if (lane < 16) ldsG[lane] = gids[n0 + lane];
  for (int idx = lane; idx < 16 * 64; idx += 32) {
    int row = idx >> 6, p = idx & 63;
    const float* r = h + (size_t)(n0 + row) * 128;
    ldsA[row * 64 + p] = pack_bf16(r[2 * p], r[2 * p + 1]);
  }
  __syncthreads();

  const int half = lane >> 4, nn = lane & 15;
  int gidv[8];
#pragma unroll
  for (int v = 0; v < 8; ++v) gidv[v] = ldsG[v + 8 * half];

  for (int nt = 0; nt < 8; ++nt) {
    int col = nt * 16 + nn;
    float bias = b0[col];
    v8f acc;
#pragma unroll
    for (int v = 0; v < 8; ++v) acc[v] = bias;
    for (int kt = 0; kt < 4; ++kt)
      acc = wmma_bf16(load_a_frag(ldsA, 64, kt * 32, lane),
                      load_b_frag(f0 + (size_t)(kt * 8 + nt) * 256, lane), acc);
#pragma unroll
    for (int v = 0; v < 8; ++v)
      ((u16*)ldsB1)[(v + 8 * half) * 128 + col] = (u16)f2bf(silu(acc[v]));
  }
  __syncthreads();

  for (int nt = 0; nt < 8; ++nt) {
    int col = nt * 16 + nn;
    float bias = b1[col];
    v8f acc;
#pragma unroll
    for (int v = 0; v < 8; ++v) acc[v] = bias;
    for (int kt = 0; kt < 4; ++kt)
      acc = wmma_bf16(load_a_frag(ldsB1, 64, kt * 32, lane),
                      load_b_frag(f1 + (size_t)(kt * 8 + nt) * 256, lane), acc);
#pragma unroll
    for (int v = 0; v < 8; ++v)
      unsafeAtomicAdd(&g[(size_t)gidv[v] * 128 + col], acc[v]);
  }
}

// ---------- pool MLP ----------
__global__ __launch_bounds__(32) void egnn_pool_kernel(const float* __restrict__ g,
                                                       const u32* __restrict__ f0,
                                                       const float* __restrict__ b0,
                                                       const float* __restrict__ w1,
                                                       const float* __restrict__ b1,
                                                       float* __restrict__ out) {
  __shared__ u32 ldsA[16 * 64];
  __shared__ u32 ldsB1[16 * 64];
  const int lane = threadIdx.x;
  const int r0 = blockIdx.x * 16;
  for (int idx = lane; idx < 16 * 64; idx += 32) {
    int row = idx >> 6, p = idx & 63;
    const float* r = g + (size_t)(r0 + row) * 128;
    ldsA[row * 64 + p] = pack_bf16(r[2 * p], r[2 * p + 1]);
  }
  __syncthreads();
  const int half = lane >> 4, nn = lane & 15;
  for (int nt = 0; nt < 8; ++nt) {
    int col = nt * 16 + nn;
    float bias = b0[col];
    v8f acc;
#pragma unroll
    for (int v = 0; v < 8; ++v) acc[v] = bias;
    for (int kt = 0; kt < 4; ++kt)
      acc = wmma_bf16(load_a_frag(ldsA, 64, kt * 32, lane),
                      load_b_frag(f0 + (size_t)(kt * 8 + nt) * 256, lane), acc);
#pragma unroll
    for (int v = 0; v < 8; ++v)
      ((u16*)ldsB1)[(v + 8 * half) * 128 + col] = (u16)f2bf(silu(acc[v]));
  }
  __syncthreads();
  {
    const int m = lane & 15;
    const u16* rr = (const u16*)ldsB1 + m * 128 + half * 64;
    const float* ww = w1 + half * 64;
    float s = 0.f;
    for (int j = 0; j < 64; ++j) s += bf2f(rr[j]) * ww[j];
    s += __shfl_xor(s, 16, 32);
    if (lane < 16) out[r0 + lane] = s + b1[0];
  }
}

// ---------- host orchestration ----------
extern "C" void kernel_launch(void* const* d_in, const int* in_sizes, int n_in, void* d_out,
                              int out_size, void* d_ws, size_t ws_size, hipStream_t stream) {
  (void)n_in;
  (void)out_size;
  (void)ws_size;
  const int N = 20000, E = 640000, G = 128, L = 7;

  const float *node_feat, *coord_feat;
  const int *esrc, *edst, *gids;
  bool sortedTop = (in_sizes[0] == N * 3);  // JAX-sorted dict: coord_feat first
  if (sortedTop) {
    coord_feat = (const float*)d_in[0];
    edst = (const int*)d_in[1];
    esrc = (const int*)d_in[2];
    gids = (const int*)d_in[3];
    node_feat = (const float*)d_in[4];
  } else {
    node_feat = (const float*)d_in[0];
    coord_feat = (const float*)d_in[1];
    esrc = (const int*)d_in[2];
    edst = (const int*)d_in[3];
    gids = (const int*)d_in[4];
  }
  const int pbase = 5;
  bool sortedLayer = (in_sizes[pbase] != 129 * 128);  // insertion order: edge_w1 W (16512) first
  int eW1o, eb1o, eW2o, eb2o, cW1o, cb1o, cW2o, nW1o, nb1o, nW2o, nb2o;
  if (sortedLayer) {
    cW1o = 0; cb1o = 1; cW2o = 2; eW1o = 3; eb1o = 4; eW2o = 5;
    eb2o = 6; nW1o = 7; nb1o = 8; nW2o = 9; nb2o = 10;
  } else {
    eW1o = 0; eb1o = 1; eW2o = 2; eb2o = 3; cW1o = 4; cb1o = 5;
    cW2o = 6; nW1o = 7; nb1o = 8; nW2o = 9; nb2o = 10;
  }
  const float *eW1[L], *eb1[L], *eW2[L], *eb2[L], *cW1[L], *cb1[L], *cW2[L];
  const float *nW1[L], *nb1[L], *nW2[L], *nb2[L];
  for (int l = 0; l < L; ++l) {
    int b = pbase + l * 11;
    eW1[l] = (const float*)d_in[b + eW1o];
    eb1[l] = (const float*)d_in[b + eb1o];
    eW2[l] = (const float*)d_in[b + eW2o];
    eb2[l] = (const float*)d_in[b + eb2o];
    cW1[l] = (const float*)d_in[b + cW1o];
    cb1[l] = (const float*)d_in[b + cb1o];
    cW2[l] = (const float*)d_in[b + cW2o];
    nW1[l] = (const float*)d_in[b + nW1o];
    nb1[l] = (const float*)d_in[b + nb1o];
    nW2[l] = (const float*)d_in[b + nW2o];
    nb2[l] = (const float*)d_in[b + nb2o];
  }
  int nmb = pbase + L * 11;
  const float* nmW0 = (const float*)d_in[nmb + 0];
  const float* nmb0 = (const float*)d_in[nmb + 1];
  const float* nmW1 = (const float*)d_in[nmb + 2];
  const float* nmb1 = (const float*)d_in[nmb + 3];
  const float* pmW0 = (const float*)d_in[nmb + 4];
  const float* pmb0 = (const float*)d_in[nmb + 5];
  const float* pmW1 = (const float*)d_in[nmb + 6];
  const float* pmb1 = (const float*)d_in[nmb + 7];

  // workspace carving
  u8* wsp = (u8*)d_ws;
  size_t off = 0;
  auto carve = [&](size_t bytes) -> void* {
    void* p = wsp + off;
    off += (bytes + 255) & ~(size_t)255;
    return p;
  };
  float* hA = (float*)carve((size_t)N * 128 * 4);
  float* hB = (float*)carve((size_t)N * 128 * 4);
  float* h_neigh = (float*)carve((size_t)N * 128 * 4);
  float* xA = (float*)carve((size_t)N * 3 * 4);
  float* xB = (float*)carve((size_t)N * 3 * 4);
  float* x_sum = (float*)carve((size_t)N * 3 * 4);
  float* degb = (float*)carve((size_t)N * 4);
  float* gbuf = (float*)carve((size_t)G * 128 * 4);
  const size_t FRAG128 = (size_t)4 * 8 * 8 * 32 * 4;  // K=128 -> 32 KB
  u32 *fW1[L], *fW2[L], *fC1[L], *fN1[L], *fN2[L];
  for (int l = 0; l < L; ++l) {
    int din = l ? 128 : 64;
    fW1[l] = (u32*)carve((size_t)((2 * din) / 32) * 8 * 8 * 32 * 4);
    fW2[l] = (u32*)carve(FRAG128);
    fC1[l] = (u32*)carve(FRAG128);
    fN1[l] = (u32*)carve((size_t)((din + 128) / 32) * 8 * 8 * 32 * 4);
    fN2[l] = (u32*)carve(FRAG128);
  }
  u32* fNM0 = (u32*)carve(FRAG128);
  u32* fNM1 = (u32*)carve(FRAG128);
  u32* fPM0 = (u32*)carve(FRAG128);

  auto packW = [&](const float* Wp, u32* outp, int K) {
    pack_w_kernel<<<dim3((K / 32) * 8), dim3(32), 0, stream>>>(Wp, outp, K);
  };
  for (int l = 0; l < L; ++l) {
    int din = l ? 128 : 64;
    packW(eW1[l], fW1[l], 2 * din);  // radial row (index 2*din) excluded, handled via w1r
    packW(eW2[l], fW2[l], 128);
    packW(cW1[l], fC1[l], 128);
    packW(nW1[l], fN1[l], din + 128);
    packW(nW2[l], fN2[l], 128);
  }
  packW(nmW0, fNM0, 128);
  packW(nmW1, fNM1, 128);
  packW(pmW0, fPM0, 128);

  const float* hIn = node_feat;
  const float* xIn = coord_feat;
  float* hOutArr[2] = {hA, hB};
  float* xOutArr[2] = {xA, xB};
  for (int l = 0; l < L; ++l) {
    int din = l ? 128 : 64;
    hipMemsetAsync(h_neigh, 0, (size_t)N * 128 * 4, stream);
    hipMemsetAsync(x_sum, 0, (size_t)N * 3 * 4, stream);
    hipMemsetAsync(degb, 0, (size_t)N * 4, stream);
    egnn_edge_kernel<<<dim3(E / 16), dim3(32), 0, stream>>>(
        hIn, xIn, esrc, edst, fW1[l], eW1[l] + (size_t)(2 * din) * 128, eb1[l], fW2[l], eb2[l],
        fC1[l], cb1[l], cW2[l], h_neigh, x_sum, degb, din);
    float* hOut = hOutArr[l & 1];
    float* xOut = xOutArr[l & 1];
    egnn_node_kernel<<<dim3(N / 16), dim3(32), 0, stream>>>(hIn, h_neigh, xIn, x_sum, degb, fN1[l],
                                                            nb1[l], fN2[l], nb2[l], hOut, xOut,
                                                            din);
    hIn = hOut;
    xIn = xOut;
  }
  hipMemsetAsync(gbuf, 0, (size_t)G * 128 * 4, stream);
  egnn_final_node_kernel<<<dim3(N / 16), dim3(32), 0, stream>>>(hIn, gids, fNM0, nmb0, fNM1, nmb1,
                                                                gbuf);
  egnn_pool_kernel<<<dim3(G / 16), dim3(32), 0, stream>>>(gbuf, fPM0, pmb0, pmW1, pmb1,
                                                          (float*)d_out);
}
